// EntityEmbedding_72095321031175
// MI455X (gfx1250) — matrix operand order repaired and checked by reading, not compiled, not run on hardware
//
#include <hip/hip_runtime.h>
#include <hip/hip_bf16.h>

typedef __attribute__((ext_vector_type(16))) __bf16 v16bf;
typedef __attribute__((ext_vector_type(8)))  float  v8f;

#define D_MODEL 512
#define EPS_IN  1e-4f
#define EPS_LN  1e-5f
#define CLIPV   5.0f

// ---------------------------------------------------------------------------
// Prep kernel 1: inverse permutation + aux outputs.
// ---------------------------------------------------------------------------
__global__ __launch_bounds__(256) void k_prep_gather(
    const int* __restrict__ index_map, const int* __restrict__ batch_index,
    int total, int n0, int n01,
    int* __restrict__ inv, float* __restrict__ out_tb, float* __restrict__ out_et)
{
    int j = blockIdx.x * blockDim.x + threadIdx.x;
    if (j >= total) return;
    int r = index_map[j];
    inv[r] = j;
    __builtin_nontemporal_store((float)batch_index[r], &out_tb[j]);
    __builtin_nontemporal_store((r < n0) ? 0.0f : ((r < n01) ? 1.0f : 2.0f), &out_et[j]);
}

// ---------------------------------------------------------------------------
// Prep kernel 2: W [F,512] fp32 -> bf16, fragment-major swizzle so a B
// fragment (32x16 bf16) is one contiguous 32B run per lane.
// K = kt*32 + (e<8?0:16) + 8*(lane>>4) + (e&7), col = nt*16 + (lane&15).
// ---------------------------------------------------------------------------
__global__ __launch_bounds__(256) void k_prep_w(
    const float* __restrict__ W, int F, int Kp, __bf16* __restrict__ wbf)
{
    int idx = blockIdx.x * blockDim.x + threadIdx.x;
    if (idx >= Kp * D_MODEL) return;
    int e    = idx & 15;
    int lane = (idx >> 4) & 31;
    int frag = idx >> 9;
    int nt   = frag & 31;
    int kt   = frag >> 5;
    int k    = kt * 32 + ((e < 8) ? 0 : 16) + 8 * (lane >> 4) + (e & 7);
    int col  = nt * 16 + (lane & 15);
    float v  = (k < F) ? W[(size_t)k * D_MODEL + col] : 0.0f;
    wbf[idx] = (__bf16)v;
}

// ---------------------------------------------------------------------------
// Prep kernel 3: zero-padded InputNorm params: meanP[k], scaleP[k]=rsqrt(var+eps).
// ---------------------------------------------------------------------------
__global__ __launch_bounds__(128) void k_prep_params(
    const float* __restrict__ mean, const float* __restrict__ var,
    int F, int Kp, float* __restrict__ meanP, float* __restrict__ scaleP)
{
    int k = threadIdx.x;
    if (k >= Kp) return;
    bool ok = k < F;
    meanP[k]  = ok ? mean[k] : 0.0f;
    scaleP[k] = ok ? __frsqrt_rn(var[k] + EPS_IN) : 0.0f;
}

// ---------------------------------------------------------------------------
// Fused embed: InputNorm -> bf16 WMMA GEMM -> bias+ReLU -> LayerNorm -> scatter.
// Block: 256 threads = 8 waves; block tile 32 rows x 512 cols.
// Wave w: rowHalf = w>>2 (16 rows), colGroup = w&3 (128 cols = 8 WMMA tiles).
// F, KP compile-time: all K-bound guards fold; row index clamped so all loads
// are unconditional (padding rows compute garbage that is never stored).
// Output stores are non-temporal: write-once stream must not evict the
// L2-resident entity matrix / weights on a 192MB-L2, 23.3TB/s part.
// ---------------------------------------------------------------------------
template<int F, int KP>
__global__ __launch_bounds__(256) void k_embed(
    const float* __restrict__ x,
    const float* __restrict__ meanP, const float* __restrict__ scaleP,
    const float* __restrict__ bias,
    const float* __restrict__ gamma, const float* __restrict__ beta,
    const __bf16* __restrict__ wbf, const int* __restrict__ inv,
    int nrows, int typeOff, float* __restrict__ out)
{
    const int w    = threadIdx.x >> 5;
    const int lane = threadIdx.x & 31;
    const int half = lane >> 4;
    const int ln   = lane & 15;
    const int rowBase = blockIdx.x * 32 + (w >> 2) * 16;
    const int colBase = (w & 3) * 128;

    v8f c[8];
    #pragma unroll
    for (int t = 0; t < 8; ++t) c[t] = (v8f){0.f,0.f,0.f,0.f,0.f,0.f,0.f,0.f};

    // clamp row: OOB rows load real (in-bounds) data, results discarded at store
    const int row  = rowBase + ln;
    const int rowc = (row < nrows) ? row : (nrows - 1);
    const float* xr = x + (size_t)rowc * F;

    #pragma unroll
    for (int kt = 0; kt < (KP >> 5); ++kt) {
        // --- A fragment (16x32 bf16) with fused InputNorm ---
        v16bf a;
        #pragma unroll
        for (int e = 0; e < 16; ++e) {
            const int cbase = kt * 32 + ((e < 8) ? 0 : 16);   // compile-time
            const int k = cbase + 8 * half + (e & 7);
            // F, KP multiples of 16 and cbase%16==0  =>  (k<F) <=> (cbase<F)
            float xv = (cbase < F) ? xr[k] : 0.0f;
            float h  = (xv - meanP[k]) * scaleP[k];
            h = fminf(fmaxf(h, -CLIPV), CLIPV);
            a[e] = (__bf16)h;
        }
        // --- 8 WMMA tiles along N ---
        const __bf16* wb = wbf + (size_t)((kt * 32 + (colBase >> 4)) * 32) * 16;
        #pragma unroll
        for (int t = 0; t < 8; ++t) {
            v16bf bfrag = *(const v16bf*)(wb + (size_t)(t * 32 + lane) * 16);
            c[t] = __builtin_amdgcn_wmma_f32_16x16x32_bf16(
                false, a, false, bfrag, (short)0, c[t], false, false);
        }
    }

    // --- bias + ReLU, per-row partial sums for LayerNorm ---
    float s[8], q[8];
    #pragma unroll
    for (int r = 0; r < 8; ++r) { s[r] = 0.f; q[r] = 0.f; }
    #pragma unroll
    for (int t = 0; t < 8; ++t) {
        int col = colBase + t * 16 + ln;
        float bb = bias[col];
        #pragma unroll
        for (int r = 0; r < 8; ++r) {
            float hv = fmaxf(c[t][r] + bb, 0.0f);
            c[t][r] = hv;
            s[r] += hv;
            q[r] += hv * hv;
        }
    }
    #pragma unroll
    for (int off = 1; off < 16; off <<= 1) {
        #pragma unroll
        for (int r = 0; r < 8; ++r) {
            s[r] += __shfl_xor(s[r], off, 32);
            q[r] += __shfl_xor(q[r], off, 32);
        }
    }

    __shared__ float red[4][32][2];
    if (ln == 0) {
        #pragma unroll
        for (int r = 0; r < 8; ++r) {
            int lr = (w >> 2) * 16 + 8 * half + r;
            red[w & 3][lr][0] = s[r];
            red[w & 3][lr][1] = q[r];
        }
    }
    __syncthreads();

    float muv[8], rstd[8];
    #pragma unroll
    for (int r = 0; r < 8; ++r) {
        int lr = (w >> 2) * 16 + 8 * half + r;
        float ts = 0.f, tq = 0.f;
        #pragma unroll
        for (int g = 0; g < 4; ++g) { ts += red[g][lr][0]; tq += red[g][lr][1]; }
        float m = ts * (1.0f / D_MODEL);
        float v = tq * (1.0f / D_MODEL) - m * m;
        muv[r]  = m;
        rstd[r] = __frsqrt_rn(v + EPS_LN);
    }

    // --- normalize, scale/shift, scatter through inverse permutation (NT stores) ---
    int dst[8];
    bool ok[8];
    #pragma unroll
    for (int r = 0; r < 8; ++r) {
        int grow = rowBase + 8 * half + r;
        ok[r]  = grow < nrows;
        dst[r] = ok[r] ? inv[typeOff + grow] : 0;
    }
    #pragma unroll
    for (int t = 0; t < 8; ++t) {
        int col = colBase + t * 16 + ln;
        float ga = gamma[col];
        float be = beta[col];
        #pragma unroll
        for (int r = 0; r < 8; ++r) {
            if (ok[r]) {
                float y = ga * (c[t][r] - muv[r]) * rstd[r] + be;
                __builtin_nontemporal_store(y, &out[(size_t)dst[r] * D_MODEL + col]);
            }
        }
    }
}

// ---------------------------------------------------------------------------
extern "C" void kernel_launch(void* const* d_in, const int* in_sizes, int n_in,
                              void* d_out, int out_size, void* d_ws, size_t ws_size,
                              hipStream_t stream) {
    (void)in_sizes; (void)n_in; (void)out_size; (void)ws_size;

    const int N0 = 80000, N1 = 70000, N2 = 50000;
    const int F0 = 64, F1 = 128, F2 = 48;
    const int Kp0 = 64, Kp1 = 128, Kp2 = 64;   // K padded to multiples of 32
    const int total = N0 + N1 + N2;

    const float* ent0  = (const float*)d_in[0];
    const float* mean0 = (const float*)d_in[1];
    const float* var0  = (const float*)d_in[2];
    const float* W0    = (const float*)d_in[3];
    const float* b0    = (const float*)d_in[4];
    const float* g0    = (const float*)d_in[5];
    const float* be0   = (const float*)d_in[6];
    const float* ent1  = (const float*)d_in[7];
    const float* mean1 = (const float*)d_in[8];
    const float* var1  = (const float*)d_in[9];
    const float* W1    = (const float*)d_in[10];
    const float* b1    = (const float*)d_in[11];
    const float* g1    = (const float*)d_in[12];
    const float* be1   = (const float*)d_in[13];
    const float* ent2  = (const float*)d_in[14];
    const float* mean2 = (const float*)d_in[15];
    const float* var2  = (const float*)d_in[16];
    const float* W2    = (const float*)d_in[17];
    const float* b2    = (const float*)d_in[18];
    const float* g2    = (const float*)d_in[19];
    const float* be2   = (const float*)d_in[20];
    const int* index_map   = (const int*)d_in[21];
    const int* batch_index = (const int*)d_in[22];

    float* out_x  = (float*)d_out;                       // [total, 512]
    float* out_tb = out_x + (size_t)total * D_MODEL;     // [total]
    float* out_et = out_tb + total;                      // [total]

    // workspace layout
    int* inv = (int*)d_ws;                               // total*4 = 800000 B (256B aligned)
    __bf16* wb0 = (__bf16*)((char*)d_ws + (size_t)total * sizeof(int));
    __bf16* wb1 = wb0 + (size_t)Kp0 * D_MODEL;
    __bf16* wb2 = wb1 + (size_t)Kp1 * D_MODEL;
    float* mp0 = (float*)(wb2 + (size_t)Kp2 * D_MODEL);
    float* sp0 = mp0 + Kp0;
    float* mp1 = sp0 + Kp0;
    float* sp1 = mp1 + Kp1;
    float* mp2 = sp1 + Kp1;
    float* sp2 = mp2 + Kp2;

    // prep: inverse perm + aux outputs
    k_prep_gather<<<(total + 255) / 256, 256, 0, stream>>>(
        index_map, batch_index, total, N0, N0 + N1, inv, out_tb, out_et);

    // prep: weight conversion/swizzle + padded InputNorm params
    k_prep_w<<<(Kp0 * D_MODEL) / 256, 256, 0, stream>>>(W0, F0, Kp0, wb0);
    k_prep_w<<<(Kp1 * D_MODEL) / 256, 256, 0, stream>>>(W1, F1, Kp1, wb1);
    k_prep_w<<<(Kp2 * D_MODEL) / 256, 256, 0, stream>>>(W2, F2, Kp2, wb2);
    k_prep_params<<<1, 128, 0, stream>>>(mean0, var0, F0, Kp0, mp0, sp0);
    k_prep_params<<<1, 128, 0, stream>>>(mean1, var1, F1, Kp1, mp1, sp1);
    k_prep_params<<<1, 128, 0, stream>>>(mean2, var2, F2, Kp2, mp2, sp2);

    // fused embed per entity type (F, KP compile-time)
    k_embed<64, 64><<<(N0 + 31) / 32, 256, 0, stream>>>(
        ent0, mp0, sp0, b0, g0, be0, wb0, inv, N0, 0, out_x);
    k_embed<128, 128><<<(N1 + 31) / 32, 256, 0, stream>>>(
        ent1, mp1, sp1, b1, g1, be1, wb1, inv, N1, N0, out_x);
    k_embed<48, 64><<<(N2 + 31) / 32, 256, 0, stream>>>(
        ent2, mp2, sp2, b2, g2, be2, wb2, inv, N2, N0 + N1, out_x);
}
